// ScaledDotProductAttention_84902913507681
// MI455X (gfx1250) — matrix operand order repaired and checked by reading, not compiled, not run on hardware
//
#include <hip/hip_runtime.h>
#include <hip/hip_bf16.h>

typedef __attribute__((ext_vector_type(16))) _Float16 v16h;
typedef __attribute__((ext_vector_type(8)))  _Float16 v8h;
typedef __attribute__((ext_vector_type(8)))  float    v8f;

#define BH      32
#define S_LEN   2048
#define DK      64
#define TQ      32            // Q rows per workgroup (halves K/V L2 traffic vs 16)
#define NW      8             // waves per workgroup (wave32)
#define PROW    (S_LEN + 8)   // f16 score/prob row stride (mult of 8 -> 16B vec loads)
#define MASK_NEG (-30000.0f)  // f16-representable "-inf": exp underflows to 0

// LDS budget: 32*2056*2 (f16 scores, overwritten in place by f16 P,
//                        later reused for fp32 O partials)   = 131584 B
//           + 32*8*4 red + 32*4 rmax + 32*4 rinv             =   1280 B
//           -> ~130 KiB => 2 workgroups per 320 KiB WGP (16 waves/WGP).
__global__ __launch_bounds__(256) void attn_fused(
    const float* __restrict__ q, const float* __restrict__ kmat,
    const float* __restrict__ vmat, const unsigned char* __restrict__ pad,
    const float* __restrict__ cmask, float* __restrict__ out_o,
    float* __restrict__ out_a)
{
  extern __shared__ char smem[];
  _Float16* ps   = (_Float16*)smem;                                   // [TQ][PROW]
  float*    red  = (float*)(smem + (size_t)TQ * PROW * sizeof(_Float16)); // [TQ][8]
  float*    rmax = red + TQ * 8;                                      // [TQ]
  float*    rinv = rmax + TQ;                                         // [TQ]

  const int tid    = threadIdx.x;
  const int wave   = tid >> 5;
  const int lane   = tid & 31;
  const int mrow16 = lane & 15;   // row (A) or column (B, C/D) within a 16x16 tile
  const int hi     = lane >> 4;   // lane half selects K-subgroup per ISA layout

  const int b  = blockIdx.y;
  const int q0 = blockIdx.x * TQ;

  // ---------- load Q tile (two 16-row tiles) into WMMA A fragments ----------
  // A 16x32 layout: lanes 0-15 hold K={d0..d0+7, d0+16..d0+23} with d0=0,
  // lanes 16-31 the same with d0=8.  Fragment c covers K=32c..32c+31 of DK=64.
  v16h aQ[2][2];
#pragma unroll
  for (int rt = 0; rt < 2; ++rt) {
    const float* qrow = q + ((size_t)b * S_LEN + q0 + rt * 16 + mrow16) * DK;
    const int d0 = hi * 8;
#pragma unroll
    for (int e = 0; e < 8; ++e) {
      aQ[rt][0][e]     = (_Float16)qrow[d0 + e];
      aQ[rt][0][8 + e] = (_Float16)qrow[d0 + 16 + e];
      aQ[rt][1][e]     = (_Float16)qrow[32 + d0 + e];
      aQ[rt][1][8 + e] = (_Float16)qrow[32 + d0 + 16 + e];
    }
  }

  // ---------- phase 1: f16 scores = (Q K^T)/sqrt(dk), pad-masked, in LDS ----------
  const unsigned char* padb = pad + ((size_t)b * S_LEN + q0) * S_LEN;
  for (int kt = wave * 16; kt < wave * 16 + 16; ++kt) {
    // B 32x16 layout: lane n (0-15) holds K rows {d0..d0+7, d0+16..d0+23} of column n
    // (d0 = 0 for lanes 0-15, 8 for lanes 16-31). B[d][n] = K[n][d].
    const float* krow = kmat + ((size_t)b * S_LEN + kt * 16 + mrow16) * DK;
    const int d0 = hi * 8;
    v16h bK0, bK1;
#pragma unroll
    for (int e = 0; e < 16; ++e) {
      bK0[e] = (_Float16)krow[d0 + e + (e & 8)];
      bK1[e] = (_Float16)krow[32 + d0 + e + (e & 8)];
    }
    const int ncol = kt * 16 + mrow16;
#pragma unroll
    for (int rt = 0; rt < 2; ++rt) {        // two 16-row tiles share bK0/bK1
      v8f c8 = {};
      c8 = __builtin_amdgcn_wmma_f32_16x16x32_f16(false, aQ[rt][0], false, bK0,
                                                  (short)0, c8, false, false);
      c8 = __builtin_amdgcn_wmma_f32_16x16x32_f16(false, aQ[rt][1], false, bK1,
                                                  (short)0, c8, false, false);
      // C/D layout: lanes 0-15: N=lane, VGPR r -> M=r; lanes 16-31: M=8+r.
#pragma unroll
      for (int r = 0; r < 8; ++r) {
        const int mr = rt * 16 + hi * 8 + r;
        float s = c8[r] * 0.125f;           // 1/sqrt(64)
        // masks are touch-once: non-temporal so they don't pollute L2
        if (__builtin_nontemporal_load(&padb[(size_t)mr * S_LEN + ncol]))
          s = MASK_NEG;
        ps[mr * PROW + ncol] = (_Float16)s;
      }
    }
  }
  __syncthreads();

  // ---------- phase 2: per-row max and exp-sum (8 threads per row) ----------
  {
    const int row = tid >> 3, j = tid & 7;
    const _Float16* srow = ps + row * PROW;
    float mx = -3.0e38f;
    for (int kk = j; kk < S_LEN; kk += 8) mx = fmaxf(mx, (float)srow[kk]);
    red[row * 8 + j] = mx;
    __syncthreads();
    mx = red[row * 8];
#pragma unroll
    for (int t = 1; t < 8; ++t) mx = fmaxf(mx, red[row * 8 + t]);
    float sum = 0.f;
    for (int kk = j; kk < S_LEN; kk += 8) sum += __expf((float)srow[kk] - mx);
    __syncthreads();                        // all reads of red done before overwrite
    red[row * 8 + j] = sum;
    __syncthreads();
    if (j == 0) {
      float tot = 0.f;
#pragma unroll
      for (int t = 0; t < 8; ++t) tot += red[row * 8 + t];
      rmax[row] = mx;
      rinv[row] = 1.0f / tot;
    }
  }
  __syncthreads();

  // ---------- phase 3: normalize * context_mask -> attention (NT) + f16 P ----------
  // Wave-local column slice; in-place per-element f16 overwrite (same thread
  // reads then writes each element), so phase 4 needs no extra barrier.
  const int c0 = wave * (S_LEN / NW);       // 256 columns per wave
  const float* cmb  = cmask + ((size_t)b * S_LEN + q0) * S_LEN;
  float*       attn = out_a + ((size_t)b * S_LEN + q0) * S_LEN;
  for (int mr = 0; mr < TQ; ++mr) {
    const float mxr = rmax[mr];
    const float inv = rinv[mr];
#pragma unroll
    for (int i = 0; i < 8; ++i) {
      const int kk = c0 + i * 32 + lane;    // coalesced 128B per instruction
      float p = __expf((float)ps[mr * PROW + kk] - mxr) * inv;
      float a = p * __builtin_nontemporal_load(&cmb[(size_t)mr * S_LEN + kk]);
      __builtin_nontemporal_store(a, &attn[(size_t)mr * S_LEN + kk]);
      ps[mr * PROW + kk] = (_Float16)a;
    }
  }

  // ---------- phase 4: O_partial = P_wave @ V_wave via WMMA ----------
  v8f acc[2][4] = {{{}, {}, {}, {}}, {{}, {}, {}, {}}};
  for (int ch = 0; ch < 8; ++ch) {          // 8 x K=32 chunks = wave's 256 cols
    const int cb = c0 + ch * 32;
    v16h aP[2];
#pragma unroll
    for (int rt = 0; rt < 2; ++rt) {        // A fragment straight from LDS
      const v8h* pa = (const v8h*)(ps + (rt * 16 + mrow16) * PROW + cb + hi * 8);
      v8h lo = pa[0];
      v8h hh = pa[2];                       // +16 halves
      aP[rt] = __builtin_shufflevector(lo, hh, 0, 1, 2, 3, 4, 5, 6, 7,
                                       8, 9, 10, 11, 12, 13, 14, 15);
    }
    const int kb = cb + hi * 8;
    const float* vbase = vmat + ((size_t)b * S_LEN + kb) * DK + mrow16;
#pragma unroll
    for (int t = 0; t < 4; ++t) {
      const float* vcol = vbase + t * 16;   // V column, stride DK (kept L2-hot)
      v16h bV;
#pragma unroll
      for (int e = 0; e < 16; ++e) bV[e] = (_Float16)vcol[(size_t)(e + (e & 8)) * DK];
#pragma unroll
      for (int rt = 0; rt < 2; ++rt)        // reuse bV across both row tiles
        acc[rt][t] = __builtin_amdgcn_wmma_f32_16x16x32_f16(
            false, aP[rt], false, bV, (short)0, acc[rt][t], false, false);
    }
  }

  // ---------- phase 5: cross-wave reduction of O, reusing score LDS ----------
  __syncthreads();                          // everyone finished reading ps
  float* part = (float*)smem;               // [NW][TQ][DK] = 64 KiB (fits in ps)
#pragma unroll
  for (int rt = 0; rt < 2; ++rt)
#pragma unroll
    for (int t = 0; t < 4; ++t)
#pragma unroll
      for (int r = 0; r < 8; ++r) {
        const int mr = rt * 16 + hi * 8 + r;
        const int dc = t * 16 + mrow16;
        part[(wave * TQ + mr) * DK + dc] = acc[rt][t][r];
      }
  __syncthreads();
  for (int idx = tid; idx < TQ * DK; idx += 256) {
    const int mr = idx >> 6, dc = idx & 63;
    float s = 0.f;
#pragma unroll
    for (int w = 0; w < NW; ++w) s += part[(w * TQ + mr) * DK + dc];
    out_o[((size_t)b * S_LEN + q0 + mr) * DK + dc] = s;
  }
}

extern "C" void kernel_launch(void* const* d_in, const int* in_sizes, int n_in,
                              void* d_out, int out_size, void* d_ws, size_t ws_size,
                              hipStream_t stream) {
  (void)in_sizes; (void)n_in; (void)out_size; (void)d_ws; (void)ws_size;
  const float*         q   = (const float*)d_in[0];
  const float*         k   = (const float*)d_in[1];
  const float*         v   = (const float*)d_in[2];
  const unsigned char* pad = (const unsigned char*)d_in[3];  // jnp.bool_ = 1 byte
  const float*         cm  = (const float*)d_in[4];

  float* out_o = (float*)d_out;                              // [BH,S,DK]
  float* out_a = out_o + (size_t)BH * S_LEN * DK;            // [BH,S,S]

  const size_t lds = (size_t)TQ * PROW * sizeof(_Float16)
                   + (size_t)(TQ * 8 + TQ + TQ) * sizeof(float);
  // Opt into >64 KiB dynamic LDS (CDNA5 WGP has 320 KiB). Deterministic,
  // not a stream op, safe under graph capture.
  hipFuncSetAttribute(reinterpret_cast<const void*>(attn_fused),
                      hipFuncAttributeMaxDynamicSharedMemorySize, (int)lds);

  dim3 grid(S_LEN / TQ, BH);                                 // 64 x 32 workgroups
  attn_fused<<<grid, 256, lds, stream>>>(q, k, v, pad, cm, out_o, out_a);
}
